// CIG_Model_2963527434636
// MI455X (gfx1250) — matrix operand order repaired and checked by reading, not compiled, not run on hardware
//
#include <hip/hip_runtime.h>
#include <hip/hip_bf16.h>
#include <cstdint>
#include <cstddef>

typedef __bf16 bf16;
typedef __attribute__((ext_vector_type(16))) __bf16 bf16x16;
typedef __attribute__((ext_vector_type(8)))  float  f32x8;

#define MODE_ENC   0   // Conv2d k3 s2 p1, input NHWC bf16
#define MODE_DEC   1   // ConvTranspose2d k3 s2 p1 op1 == conv over 2x-dilated input
#define MODE_DENSE 2   // plain row-major GEMM

// ---- CDNA5 async global->LDS staging (guarded: falls back to sync copy) ----
#if defined(__AMDGCN__) && __has_builtin(__builtin_amdgcn_global_load_async_to_lds_b128)
#define HAVE_ASYNC_LDS 1
#else
#define HAVE_ASYNC_LDS 0
#endif
#if defined(__AMDGCN__) && __has_builtin(__builtin_amdgcn_s_wait_asynccnt)
#define HAVE_WAIT_ASYNC 1
#else
#define HAVE_WAIT_ASYNC 0
#endif

// Builtin signature (from toolchain diagnostic): takes pointers to
// 'int __attribute__((vector_size(16)))' in the generic address space.
typedef int v4i_vs __attribute__((vector_size(16)));

__device__ __forceinline__ void wait_async_lds()
{
#if HAVE_ASYNC_LDS
#if HAVE_WAIT_ASYNC
  __builtin_amdgcn_s_wait_asynccnt(0);
#else
  asm volatile("s_wait_asynccnt 0" ::: "memory");
#endif
#endif
}

// ---------------------------------------------------------------------------
// Unified implicit-GEMM kernel, bf16 WMMA, f32 accumulate.
//   C[M,N] = A(im2col / dense)[M,K] * B[K,N] (+bias)
// Block: 256 threads = 8 waves. BM=128, BN=64, KB=32.
// Each wave computes a 16x64 strip as 4 x v_wmma_f32_16x16x32_bf16.
// cinShift = log2(Cin) when Cin is a power of two (>=16): fast tap decode +
// vectorized b128 A loads (all 16 k's of a thread share one (ky,kx) tap).
// ---------------------------------------------------------------------------
__global__ __launch_bounds__(256)
void wmma_igemm(const bf16* __restrict__ A, const bf16* __restrict__ Bw,
                const float* __restrict__ bias, float* __restrict__ Y,
                int mode, int bKmajor, int hasBias, int cinShift,
                int Mtot, int Ntot, int Kdim, int lda, int ldb, int ldc,
                int Hin, int Win, int Cin, int Hout, int Wout, int Cout)
{
  constexpr int BM = 128, BN = 64, KB = 32;
  __shared__ __align__(16) bf16 As[BM][KB];
  __shared__ __align__(16) bf16 Bs[BN][KB];

  const int tid  = threadIdx.x;
  const int lane = tid & 31;
  const int wave = tid >> 5;
  const int m0 = blockIdx.x * BM;
  const int n0 = blockIdx.y * BN;
  const int HW = Hout * Wout;

  f32x8 acc[4] = {};

  const int Kpad = (Kdim + KB - 1) & ~(KB - 1);

  // per-thread A-loader constants: thread loads row (tid>>1), k-half (tid&1)
  const int ar    = tid >> 1;
  const int ahalf = tid & 1;
  const int am    = m0 + ar;
  const bool amok = (am < Mtot);
  int ab = 0, aoy = 0, aox = 0;
  if (mode != MODE_DENSE && amok) {
    ab = am / HW;
    int rem = am - ab * HW;
    aoy = rem / Wout;
    aox = rem - aoy * Wout;
  }

  for (int k0 = 0; k0 < Kpad; k0 += KB) {
    __syncthreads();

    // ---- stage A tile: im2col gather (conv/deconv) or dense row ----
    if (mode == MODE_DENSE) {
      // dense uses have Kdim % 32 == 0 -> no k guard
      const int kbase = k0 + ahalf * 16;
      uint4 a0 = {0, 0, 0, 0}, a1 = {0, 0, 0, 0};
      if (amok) {
        const bf16* src = A + (size_t)am * lda + kbase;
        a0 = *(const uint4*)src;
        a1 = *(const uint4*)(src + 8);
      }
      *(uint4*)&As[ar][ahalf * 16]     = a0;
      *(uint4*)&As[ar][ahalf * 16 + 8] = a1;
    } else if (cinShift >= 4) {
      // Cin = 2^cinShift (>=16): 16 contiguous k share one tap, contiguous ci
      const int kbase = k0 + ahalf * 16;
      const int tap = kbase >> cinShift;      // 0..8 (Kdim = Cin*9, mult of 32)
      const int ci0 = kbase & ((1 << cinShift) - 1);
      const int ky = tap / 3;
      const int kx = tap - ky * 3;
      const bf16* src = nullptr;
      bool ok = false;
      if (amok) {
        if (mode == MODE_ENC) {
          const int iy = aoy * 2 - 1 + ky;
          const int ix = aox * 2 - 1 + kx;
          ok = (iy >= 0 && iy < Hin && ix >= 0 && ix < Win);
          if (ok) src = A + (((size_t)ab * Hin + iy) * Win + ix) * Cin + ci0;
        } else {
          const int dy = aoy + ky - 1;
          const int dx = aox + kx - 1;
          if (dy >= 0 && dx >= 0 && !(dy & 1) && !(dx & 1)) {
            const int iy = dy >> 1, ix = dx >> 1;
            ok = (iy < Hin && ix < Win);
            if (ok) src = A + (((size_t)ab * Hin + iy) * Win + ix) * Cin + ci0;
          }
        }
      }
      uint4 a0 = {0, 0, 0, 0}, a1 = {0, 0, 0, 0};
      if (ok) { a0 = *(const uint4*)src; a1 = *(const uint4*)(src + 8); }
      *(uint4*)&As[ar][ahalf * 16]     = a0;
      *(uint4*)&As[ar][ahalf * 16 + 8] = a1;
    } else {
      // generic scalar path (conv1: Cin=3, Kdim=27)
      #pragma unroll
      for (int j = 0; j < 16; ++j) {
        const int kl = ahalf * 16 + j;
        const int kk = k0 + kl;
        bf16 v = (bf16)0.0f;
        if (amok && kk < Kdim) {
          const int tap = kk / Cin;
          const int ci  = kk - tap * Cin;
          const int ky  = tap / 3;
          const int kx  = tap - ky * 3;
          if (mode == MODE_ENC) {
            const int iy = aoy * 2 - 1 + ky;
            const int ix = aox * 2 - 1 + kx;
            if (iy >= 0 && iy < Hin && ix >= 0 && ix < Win)
              v = A[(((size_t)ab * Hin + iy) * Win + ix) * Cin + ci];
          } else {
            const int dy = aoy + ky - 1;
            const int dx = aox + kx - 1;
            if (dy >= 0 && dx >= 0 && !(dy & 1) && !(dx & 1)) {
              const int iy = dy >> 1, ix = dx >> 1;
              if (iy < Hin && ix < Win)
                v = A[(((size_t)ab * Hin + iy) * Win + ix) * Cin + ci];
            }
          }
        }
        As[ar][kl] = v;
      }
    }

    // ---- stage B tile into LDS as [n][k] (N-major rows of 32 bf16) ----
    if (!bKmajor) {
      // B global is [Npad][Kpad] (rows padded with zeros) -> contiguous b128s
      const int n = tid >> 2;
      const int q = tid & 3;
      const bf16* src = Bw + (size_t)(n0 + n) * ldb + k0 + q * 8;
#if HAVE_ASYNC_LDS
      __builtin_amdgcn_global_load_async_to_lds_b128(
          (v4i_vs*)src, (v4i_vs*)&Bs[n][q * 8], 0, 0);
#else
      *(uint4*)&Bs[n][q * 8] = *(const uint4*)src;
#endif
      if (k0 + KB < Kpad) __builtin_prefetch(src + KB, 0, 0);
    } else {
      // B global is [K][Nld] row-major (N a multiple of 64 in our uses)
      const int k  = tid >> 3;
      const int n8 = (tid & 7) * 8;
      const bf16* src = Bw + (size_t)(k0 + k) * ldb + n0 + n8;
      uint4 t4 = *(const uint4*)src;
      const bf16* tv = (const bf16*)&t4;
      #pragma unroll
      for (int e = 0; e < 8; ++e) Bs[n8 + e][k] = tv[e];
      if (k0 + KB < Kpad) __builtin_prefetch(src + (size_t)KB * ldb, 0, 0);
    }
    wait_async_lds();
    __syncthreads();

    // ---- build fragments (k = 16*half + s in both A and B), issue 4 WMMAs ----
    {
      const int mrow = lane & 15;
      const int half = lane >> 4;                 // lane half selects K sub-range
      union { bf16x16 v; unsigned u[8]; } af;
      const unsigned* arow = (const unsigned*)&As[wave * 16 + mrow][0];
      #pragma unroll
      for (int r = 0; r < 8; ++r) {
        // VGPR r holds K pair: k=2r + half*16
        const int pair = half * 8 + r;
        af.u[r] = arow[pair];
      }
      const int nl = lane & 15;
      #pragma unroll
      for (int t = 0; t < 4; ++t) {
        union { bf16x16 v; unsigned u[8]; } bfv;
        const unsigned* brow = (const unsigned*)&Bs[t * 16 + nl][0];
        #pragma unroll
        for (int r = 0; r < 8; ++r) bfv.u[r] = brow[half * 8 + r];
        acc[t] = __builtin_amdgcn_wmma_f32_16x16x32_bf16(
            false, af.v, false, bfv.v, (short)0, acc[t], false, false);
      }
    }
  }

  // ---- epilogue: C layout VGPR j -> (M=j | M=8+j), N = lane%16 ----
  const int half = lane >> 4;
  const int nl   = lane & 15;
  float bv[4];
  #pragma unroll
  for (int t = 0; t < 4; ++t) {
    const int n = n0 + t * 16 + nl;
    bv[t] = (hasBias && n < Ntot) ? bias[n] : 0.0f;
  }
  #pragma unroll
  for (int j = 0; j < 8; ++j) {
    const int m = m0 + wave * 16 + half * 8 + j;
    if (m >= Mtot) continue;
    size_t rowbase;
    if (mode == MODE_DENSE) {
      rowbase = (size_t)m * ldc;
    } else {
      const int b = m / HW;
      const int rem = m - b * HW;
      const int oy = rem / Wout;
      const int ox = rem - oy * Wout;
      rowbase = (((size_t)b * Hout + oy) * Wout + ox) * Cout;
    }
    #pragma unroll
    for (int t = 0; t < 4; ++t) {
      const int n = n0 + t * 16 + nl;
      if (n < Ntot) Y[rowbase + n] = acc[t][j] + bv[t];
    }
  }
}

// ---------------------------------------------------------------------------
// Support kernels
// ---------------------------------------------------------------------------
__global__ void k_nchw_to_nhwc_bf16(const float* __restrict__ in, bf16* __restrict__ out,
                                    int B, int C, int H, int W)
{
  size_t i = (size_t)blockIdx.x * blockDim.x + threadIdx.x;
  size_t total = (size_t)B * C * H * W;
  if (i >= total) return;
  int x = (int)(i % W); size_t r = i / W;
  int y = (int)(r % H); r /= H;
  int c = (int)(r % C); int b = (int)(r / C);
  out[(((size_t)b * H + y) * W + x) * C + c] = (bf16)in[i];
}

__global__ void k_nhwc_to_nchw(const float* __restrict__ in, float* __restrict__ out,
                               int B, int C, int H, int W)
{
  size_t i = (size_t)blockIdx.x * blockDim.x + threadIdx.x;
  size_t total = (size_t)B * C * H * W;
  if (i >= total) return;
  int x = (int)(i % W); size_t r = i / W;
  int y = (int)(r % H); r /= H;
  int c = (int)(r % C); int b = (int)(r / C);
  out[i] = in[(((size_t)b * H + y) * W + x) * C + c];
}

// Conv2d weight (Cout,Cin,3,3) f32 -> [Npad][Kpad] bf16, k = (ky*3+kx)*Cin + ci
__global__ void k_prep_wenc(const float* __restrict__ w, bf16* __restrict__ Wt,
                            int Cout, int Cin, int Kpad, int Npad)
{
  size_t i = (size_t)blockIdx.x * blockDim.x + threadIdx.x;
  if (i >= (size_t)Npad * Kpad) return;
  int k = (int)(i % Kpad);
  int n = (int)(i / Kpad);
  bf16 v = (bf16)0.0f;
  const int Kdim = Cin * 9;
  if (n < Cout && k < Kdim) {
    int tap = k / Cin, ci = k - tap * Cin;
    int ky = tap / 3, kx = tap - ky * 3;
    v = (bf16)w[(((size_t)n * Cin + ci) * 3 + ky) * 3 + kx];
  }
  Wt[i] = v;
}

// ConvTranspose2d weight (Cin,Cout,3,3) torch layout -> flipped+transposed [Npad][Kpad]
__global__ void k_prep_wdec(const float* __restrict__ w, bf16* __restrict__ Wt,
                            int Cout, int Cin, int Kpad, int Npad)
{
  size_t i = (size_t)blockIdx.x * blockDim.x + threadIdx.x;
  if (i >= (size_t)Npad * Kpad) return;
  int k = (int)(i % Kpad);
  int n = (int)(i / Kpad);
  bf16 v = (bf16)0.0f;
  const int Kdim = Cin * 9;
  if (n < Cout && k < Kdim) {
    int tap = k / Cin, ci = k - tap * Cin;
    int ky = tap / 3, kx = tap - ky * 3;
    v = (bf16)w[(((size_t)ci * Cout + n) * 3 + (2 - ky)) * 3 + (2 - kx)];
  }
  Wt[i] = v;
}

__global__ void k_f32_to_bf16(const float* __restrict__ in, bf16* __restrict__ out, size_t total)
{
  size_t i = (size_t)blockIdx.x * blockDim.x + threadIdx.x;
  if (i < total) out[i] = (bf16)in[i];
}

__global__ __launch_bounds__(256)
void k_bn_stats(const float* __restrict__ Y, float* __restrict__ stats, int C, int P)
{
  const int c = blockIdx.x;
  float s = 0.f, q = 0.f;
  for (int p = threadIdx.x; p < P; p += 256) {
    float v = Y[(size_t)p * C + c];
    s += v; q += v * v;
  }
  __shared__ float rs[256], rq[256];
  rs[threadIdx.x] = s; rq[threadIdx.x] = q;
  __syncthreads();
  for (int o = 128; o > 0; o >>= 1) {
    if (threadIdx.x < o) {
      rs[threadIdx.x] += rs[threadIdx.x + o];
      rq[threadIdx.x] += rq[threadIdx.x + o];
    }
    __syncthreads();
  }
  if (threadIdx.x == 0) {
    float m = rs[0] / (float)P;
    float v = rq[0] / (float)P - m * m;
    stats[2 * c]     = m;
    stats[2 * c + 1] = rsqrtf(v + 1e-5f);
  }
}

__global__ void k_bn_apply_relu(const float* __restrict__ Y, const float* __restrict__ stats,
                                const float* __restrict__ g, const float* __restrict__ bt,
                                bf16* __restrict__ out, size_t total, int C)
{
  size_t i = (size_t)blockIdx.x * blockDim.x + threadIdx.x;
  if (i >= total) return;
  int c = (int)(i % C);
  float m = stats[2 * c], r = stats[2 * c + 1];
  float v = g[c] * (Y[i] - m) * r + bt[c];
  out[i] = (bf16)fmaxf(v, 0.0f);
}

// h (B,7,7,512) NHWC bf16 -> z (B,25088), feature index = c*49 + y*7 + x (NCHW flatten)
__global__ void k_flatten_z(const bf16* __restrict__ h, bf16* __restrict__ z, int B)
{
  size_t i = (size_t)blockIdx.x * blockDim.x + threadIdx.x;
  size_t total = (size_t)B * 25088;
  if (i >= total) return;
  int f = (int)(i % 25088); int b = (int)(i / 25088);
  int c = f / 49; int r = f - c * 49; int y = r / 7; int x = r - y * 7;
  z[i] = h[(((size_t)b * 7 + y) * 7 + x) * 512 + c];
}

__global__ void k_zhat_to_nhwc(const float* __restrict__ z, bf16* __restrict__ h, int B)
{
  size_t i = (size_t)blockIdx.x * blockDim.x + threadIdx.x;
  size_t total = (size_t)B * 25088;
  if (i >= total) return;
  int f = (int)(i % 25088); int b = (int)(i / 25088);
  int c = f / 49; int r = f - c * 49; int y = r / 7; int x = r - y * 7;
  h[(((size_t)b * 7 + y) * 7 + x) * 512 + c] = (bf16)z[i];
}

__global__ __launch_bounds__(256)
void k_rownorm_f32(const float* __restrict__ M, float* __restrict__ out, int len)
{
  const int row = blockIdx.x;
  const float* p = M + (size_t)row * len;
  float s = 0.f;
  for (int i = threadIdx.x; i < len; i += 256) { float v = p[i]; s += v * v; }
  __shared__ float rs[256];
  rs[threadIdx.x] = s; __syncthreads();
  for (int o = 128; o > 0; o >>= 1) {
    if (threadIdx.x < o) rs[threadIdx.x] += rs[threadIdx.x + o];
    __syncthreads();
  }
  if (threadIdx.x == 0) out[row] = sqrtf(rs[0]);
}

__global__ __launch_bounds__(256)
void k_rownorm_bf16(const bf16* __restrict__ M, float* __restrict__ out, int len)
{
  const int row = blockIdx.x;
  const bf16* p = M + (size_t)row * len;
  float s = 0.f;
  for (int i = threadIdx.x; i < len; i += 256) { float v = (float)p[i]; s += v * v; }
  __shared__ float rs[256];
  rs[threadIdx.x] = s; __syncthreads();
  for (int o = 128; o > 0; o >>= 1) {
    if (threadIdx.x < o) rs[threadIdx.x] += rs[threadIdx.x + o];
    __syncthreads();
  }
  if (threadIdx.x == 0) out[row] = sqrtf(rs[0]);
}

// cosine-sim scale -> softmax -> hard shrink -> L1 renorm; one block per batch row
__global__ __launch_bounds__(256)
void k_softmax_shrink(const float* __restrict__ sim, const float* __restrict__ zn,
                      const float* __restrict__ mn, float* __restrict__ wout,
                      bf16* __restrict__ wbf, int Ncols)
{
  const int b   = blockIdx.x;
  const int tid = threadIdx.x;
  __shared__ float red[256];
  constexpr int PER = 8; // Ncols == 2048
  float sv[PER];
  const float z = zn[b];
  float mx = -3.0e38f;
  #pragma unroll
  for (int j = 0; j < PER; ++j) {
    int n = tid + j * 256;
    float d = fmaxf(z * mn[n], 1e-8f);
    sv[j] = sim[(size_t)b * Ncols + n] / d;
    mx = fmaxf(mx, sv[j]);
  }
  red[tid] = mx; __syncthreads();
  for (int o = 128; o > 0; o >>= 1) { if (tid < o) red[tid] = fmaxf(red[tid], red[tid + o]); __syncthreads(); }
  mx = red[0]; __syncthreads();

  float s = 0.f;
  #pragma unroll
  for (int j = 0; j < PER; ++j) { sv[j] = __expf(sv[j] - mx); s += sv[j]; }
  red[tid] = s; __syncthreads();
  for (int o = 128; o > 0; o >>= 1) { if (tid < o) red[tid] += red[tid + o]; __syncthreads(); }
  s = red[0]; __syncthreads();

  const float t = 1.0f / (float)Ncols;
  float s2 = 0.f;
  #pragma unroll
  for (int j = 0; j < PER; ++j) {
    float w0 = sv[j] / s;
    float wi = fmaxf(w0 - t, 0.f) * w0 / (fabsf(w0 - t) + 1e-15f);
    sv[j] = wi;
    s2 += fabsf(wi);
  }
  red[tid] = s2; __syncthreads();
  for (int o = 128; o > 0; o >>= 1) { if (tid < o) red[tid] += red[tid + o]; __syncthreads(); }
  s2 = red[0];

  #pragma unroll
  for (int j = 0; j < PER; ++j) {
    int n = tid + j * 256;
    float w = sv[j] / s2;
    wout[(size_t)b * Ncols + n] = w;
    wbf [(size_t)b * Ncols + n] = (bf16)w;
  }
}

__global__ void k_zero(float* __restrict__ p, int n)
{
  int i = blockIdx.x * blockDim.x + threadIdx.x;
  if (i < n) p[i] = 0.0f;
}

// ---------------------------------------------------------------------------
// Host orchestration
// ---------------------------------------------------------------------------
static inline unsigned cdiv(size_t a, size_t b) { return (unsigned)((a + b - 1) / b); }
static inline int log2_or_neg1(int c) { return (c & (c - 1)) ? -1 : __builtin_ctz(c); }

extern "C" void kernel_launch(void* const* d_in, const int* in_sizes, int n_in,
                              void* d_out, int out_size, void* d_ws, size_t ws_size,
                              hipStream_t stream)
{
  (void)in_sizes; (void)n_in; (void)out_size; (void)ws_size;
  const int B = 16;
  const float* x = (const float*)d_in[0];
  const float *enc_w[6], *enc_b[6], *enc_g[6], *enc_bt[6];
  const float *dec_w[6], *dec_b[6], *dec_g[5], *dec_bt[5];
  for (int i = 0; i < 6; ++i) {
    enc_w[i]  = (const float*)d_in[1 + i];
    enc_b[i]  = (const float*)d_in[7 + i];
    enc_g[i]  = (const float*)d_in[13 + i];
    enc_bt[i] = (const float*)d_in[19 + i];
    dec_w[i]  = (const float*)d_in[25 + i];
    dec_b[i]  = (const float*)d_in[31 + i];
  }
  for (int i = 0; i < 5; ++i) {
    dec_g[i]  = (const float*)d_in[37 + i];
    dec_bt[i] = (const float*)d_in[42 + i];
  }
  const float* memf = (const float*)d_in[47];

  // ---- workspace carve ----
  char* ws = (char*)d_ws;
  size_t off = 0;
  auto carve = [&](size_t bytes) -> void* {
    void* p = ws + off;
    off += (bytes + 255) & ~(size_t)255;
    return p;
  };
  const size_t MAXACT = (size_t)16 * 224 * 224 * 64;  // 51,380,224
  float* Yf    = (float*)carve(MAXACT * sizeof(float));
  bf16*  actA  = (bf16*) carve(MAXACT * sizeof(bf16));
  bf16*  actB  = (bf16*) carve(MAXACT * sizeof(bf16));
  bf16*  membf = (bf16*) carve((size_t)2048 * 25088 * sizeof(bf16));
  bf16*  zbf   = (bf16*) carve((size_t)16 * 25088 * sizeof(bf16));
  bf16*  wbf   = (bf16*) carve((size_t)16 * 2048 * sizeof(bf16));
  float* simb  = (float*)carve((size_t)16 * 2048 * sizeof(float));
  float* zhat  = (float*)carve((size_t)16 * 25088 * sizeof(float));
  float* zn    = (float*)carve(16 * sizeof(float));
  float* mn    = (float*)carve(2048 * sizeof(float));
  float* stats = (float*)carve(1024 * sizeof(float));

  const int ENC_CH[7] = {3, 64, 128, 128, 256, 256, 512};
  const int DEC_CH[7] = {512, 256, 256, 128, 128, 64, 3};
  bf16* WtE[6]; bf16* WtD[6];
  int KpadE[6], NpadE[6], KpadD[6], NpadD[6];
  for (int i = 0; i < 6; ++i) {
    int K = ENC_CH[i] * 9, N = ENC_CH[i + 1];
    KpadE[i] = (K + 31) & ~31; NpadE[i] = (N + 63) & ~63;
    WtE[i] = (bf16*)carve((size_t)KpadE[i] * NpadE[i] * sizeof(bf16));
  }
  for (int i = 0; i < 6; ++i) {
    int K = DEC_CH[i] * 9, N = DEC_CH[i + 1];
    KpadD[i] = (K + 31) & ~31; NpadD[i] = (N + 63) & ~63;
    WtD[i] = (bf16*)carve((size_t)KpadD[i] * NpadD[i] * sizeof(bf16));
  }

  // ---- weight / input prep ----
  for (int i = 0; i < 6; ++i) {
    size_t tot = (size_t)KpadE[i] * NpadE[i];
    k_prep_wenc<<<cdiv(tot, 256), 256, 0, stream>>>(enc_w[i], WtE[i],
        ENC_CH[i + 1], ENC_CH[i], KpadE[i], NpadE[i]);
  }
  for (int i = 0; i < 6; ++i) {
    size_t tot = (size_t)KpadD[i] * NpadD[i];
    k_prep_wdec<<<cdiv(tot, 256), 256, 0, stream>>>(dec_w[i], WtD[i],
        DEC_CH[i + 1], DEC_CH[i], KpadD[i], NpadD[i]);
  }
  {
    size_t tot = (size_t)2048 * 25088;
    k_f32_to_bf16<<<cdiv(tot, 256), 256, 0, stream>>>(memf, membf, tot);
  }
  {
    size_t tot = (size_t)B * 3 * 448 * 448;
    k_nchw_to_nhwc_bf16<<<cdiv(tot, 256), 256, 0, stream>>>(x, actA, B, 3, 448, 448);
  }

  // ---- encoder: 6 x (WMMA implicit-GEMM conv -> BN stats -> BN+ReLU -> bf16) ----
  int H = 448;
  bf16* cur = actA; bf16* nxt = actB;
  for (int i = 0; i < 6; ++i) {
    const int Cin = ENC_CH[i], Cout = ENC_CH[i + 1];
    const int Ho = H / 2;
    const int M = B * Ho * Ho;
    dim3 g(cdiv(M, 128), (unsigned)(NpadE[i] / 64));
    wmma_igemm<<<g, 256, 0, stream>>>(cur, WtE[i], enc_b[i], Yf,
        MODE_ENC, 0, 1, log2_or_neg1(Cin), M, Cout, Cin * 9, 0, KpadE[i], 0,
        H, H, Cin, Ho, Ho, Cout);
    const int P = B * Ho * Ho;
    k_bn_stats<<<Cout, 256, 0, stream>>>(Yf, stats, Cout, P);
    size_t tot = (size_t)P * Cout;
    k_bn_apply_relu<<<cdiv(tot, 256), 256, 0, stream>>>(Yf, stats, enc_g[i], enc_bt[i], nxt, tot, Cout);
    bf16* t = cur; cur = nxt; nxt = t;
    H = Ho;
  }

  // ---- memory addressing ----
  k_flatten_z<<<cdiv((size_t)B * 25088, 256), 256, 0, stream>>>(cur, zbf, B);
  k_rownorm_bf16<<<16, 256, 0, stream>>>(zbf, zn, 25088);
  k_rownorm_f32 <<<2048, 256, 0, stream>>>(memf, mn, 25088);
  {
    // sim = z @ memory^T : M=16, N=2048, K=25088; B matrix N-major = membf[n][k]
    dim3 g(1, 2048 / 64);
    wmma_igemm<<<g, 256, 0, stream>>>(zbf, membf, nullptr, simb,
        MODE_DENSE, 0, 0, -1, 16, 2048, 25088, 25088, 25088, 2048,
        0, 0, 0, 0, 0, 0);
  }
  float* out = (float*)d_out;
  const size_t REC = (size_t)B * 3 * 448 * 448;
  float* w_out = out + REC + 160;   // [rec_x | logit_x | w]
  k_softmax_shrink<<<16, 256, 0, stream>>>(simb, zn, mn, w_out, wbf, 2048);
  {
    // z_hat = w @ memory : M=16, N=25088, K=2048; B K-major = membf[k][n]
    dim3 g(1, 25088 / 64);
    wmma_igemm<<<g, 256, 0, stream>>>(wbf, membf, nullptr, zhat,
        MODE_DENSE, 1, 0, -1, 16, 25088, 2048, 2048, 25088, 25088,
        0, 0, 0, 0, 0, 0);
  }
  k_zhat_to_nhwc<<<cdiv((size_t)B * 25088, 256), 256, 0, stream>>>(zhat, cur, B);

  // ---- decoder: 5 x (WMMA deconv -> BN+ReLU) + final deconv ----
  H = 7;
  for (int i = 0; i < 6; ++i) {
    const int Cin = DEC_CH[i], Cout = DEC_CH[i + 1];
    const int Ho = H * 2;
    const int M = B * Ho * Ho;
    dim3 g(cdiv(M, 128), (unsigned)(NpadD[i] / 64));
    wmma_igemm<<<g, 256, 0, stream>>>(cur, WtD[i], dec_b[i], Yf,
        MODE_DEC, 0, 1, log2_or_neg1(Cin), M, Cout, Cin * 9, 0, KpadD[i], 0,
        H, H, Cin, Ho, Ho, Cout);
    if (i < 5) {
      const int P = B * Ho * Ho;
      k_bn_stats<<<Cout, 256, 0, stream>>>(Yf, stats, Cout, P);
      size_t tot = (size_t)P * Cout;
      k_bn_apply_relu<<<cdiv(tot, 256), 256, 0, stream>>>(Yf, stats, dec_g[i], dec_bt[i], nxt, tot, Cout);
      bf16* t = cur; cur = nxt; nxt = t;
    }
    H = Ho;
  }

  // Yf = rec_x in NHWC (B,448,448,3); emit NCHW + zero logits
  k_nhwc_to_nchw<<<cdiv(REC, 256), 256, 0, stream>>>(Yf, out, B, 3, 448, 448);
  k_zero<<<1, 256, 0, stream>>>(out + REC, 160);
}
